// MultiHeads_91182155694377
// MI455X (gfx1250) — matrix operand order repaired and checked by reading, not compiled, not run on hardware
//
#include <hip/hip_runtime.h>
#include <hip/hip_bf16.h>
#include <math.h>

typedef _Float16 f16;
typedef __attribute__((ext_vector_type(16))) _Float16 v16h;
typedef __attribute__((ext_vector_type(8)))  float    v8f;

static constexpr int Bdim = 2;
static constexpr int Nseq = 2048;
static constexpr int Emb  = 1024;
static constexpr int Hh   = 16;
static constexpr int Dh   = 64;

// ---------------------------------------------------------------- helpers ---

static __device__ inline v8f wmma_f16(v16h a, v16h b, v8f c) {
  // D = A(16x32 f16) * B(32x16 f16) + C(16x16 f32)
  return __builtin_amdgcn_wmma_f32_16x16x32_f16(
      /*neg_a=*/false, a, /*neg_b=*/false, b,
      /*c_mod=*/(short)0, c, /*reuse_a=*/false, /*reuse_b=*/false);
}

// A-matrix 16x32 (MxK) fragment from a row-major source: src[m*ld + k].
// Lanes 0-15 hold M=lane, K=0..15; lanes 16-31 K=16..31 (matches load_bfrag).
static __device__ inline v16h load_afrag(const f16* src, int ld, int lane) {
  int m  = lane & 15;
  int kb = (lane >> 4) << 4;          // 0 or 16
  const f16* p = src + (size_t)m * ld;
  union { v16h v; uint4 q[2]; } u;
  u.q[0] = *(const uint4*)(p + kb);
  u.q[1] = *(const uint4*)(p + 8 + kb);
  return u.v;
}

// B-matrix 32x16 (KxN) fragment from a source laid out so that each column's
// K-values are contiguous: src[n*ld + k]. Lanes 0-15: N=lane, K=0..15;
// lanes 16-31: N=lane-16, K=16..31.
static __device__ inline v16h load_bfrag(const f16* src, int ld, int lane) {
  int n    = lane & 15;
  int koff = (lane >> 4) << 4;        // 0 or 16
  const f16* p = src + (size_t)n * ld + koff;
  union { v16h v; uint4 q[2]; } u;
  u.q[0] = *(const uint4*)(p);
  u.q[1] = *(const uint4*)(p + 8);
  return u.v;
}

static __device__ inline float rowmax16(float v) {
#pragma unroll
  for (int m = 1; m < 16; m <<= 1) v = fmaxf(v, __shfl_xor(v, m, 32));
  return v;
}
static __device__ inline float rowsum16(float v) {
#pragma unroll
  for (int m = 1; m < 16; m <<= 1) v += __shfl_xor(v, m, 32);
  return v;
}

// Wave-private LDS producer->consumer fence (cross-lane, same wave).
static __device__ inline void wave_lds_fence() {
  asm volatile("s_wait_dscnt 0" ::: "memory");
  __builtin_amdgcn_wave_barrier();
}

// Async global->LDS 16B copy (CDNA5 GLOBAL_LOAD_ASYNC_TO_LDS_B128, ASYNCcnt).
// Generic pointers to __shared__ carry the LDS offset in their low 32 bits.
static __device__ inline void async_copy_b128(f16* lds_dst, const f16* gsrc) {
  unsigned lds_off = (unsigned)(uintptr_t)lds_dst;
  asm volatile("global_load_async_to_lds_b128 %0, %1, off"
               :: "v"(lds_off), "v"(gsrc) : "memory");
}
static __device__ inline void async_wait0() {
  asm volatile("s_wait_asynccnt 0x0" ::: "memory");
}

// ------------------------------------------------- 1) LayerNorm -> f16 ------

__global__ __launch_bounds__(256)
void ln3_kernel(const float* __restrict__ Q, const float* __restrict__ K,
                const float* __restrict__ V, const float* __restrict__ gamma,
                const float* __restrict__ beta,
                f16* __restrict__ lnQ, f16* __restrict__ lnK, f16* __restrict__ lnV) {
  __shared__ float red1[256];
  __shared__ float red2[256];
  __shared__ float stat[2];
  const int tid = threadIdx.x;
  const int row = blockIdx.x;
  const int which = blockIdx.y;
  const float* x = (which == 0 ? Q : which == 1 ? K : V) + (size_t)row * Emb;
  f16* o = (which == 0 ? lnQ : which == 1 ? lnK : lnV) + (size_t)row * Emb;

  float4 xv = ((const float4*)x)[tid];
  red1[tid] = xv.x + xv.y + xv.z + xv.w;
  red2[tid] = xv.x*xv.x + xv.y*xv.y + xv.z*xv.z + xv.w*xv.w;
  __syncthreads();
#pragma unroll
  for (int off = 128; off > 0; off >>= 1) {
    if (tid < off) { red1[tid] += red1[tid + off]; red2[tid] += red2[tid + off]; }
    __syncthreads();
  }
  if (tid == 0) {
    float mean = red1[0] * (1.0f / Emb);
    float var  = red2[0] * (1.0f / Emb) - mean * mean;
    stat[0] = mean;
    stat[1] = rsqrtf(var + 1e-5f);
  }
  __syncthreads();
  const float mean = stat[0], rinv = stat[1];
  float4 g = ((const float4*)gamma)[tid];
  float4 bt = ((const float4*)beta)[tid];
  union { f16 h[4]; uint2 u; } w;
  w.h[0] = (f16)((xv.x - mean) * rinv * g.x + bt.x);
  w.h[1] = (f16)((xv.y - mean) * rinv * g.y + bt.y);
  w.h[2] = (f16)((xv.z - mean) * rinv * g.z + bt.z);
  w.h[3] = (f16)((xv.w - mean) * rinv * g.w + bt.w);
  ((uint2*)o)[tid] = w.u;
}

// ------------------------------------------------- 2) weights fp32 -> f16 ---

__global__ __launch_bounds__(256)
void cvtw_kernel(const float* __restrict__ Wq, const float* __restrict__ Wk,
                 const float* __restrict__ Wv,
                 f16* __restrict__ oq, f16* __restrict__ ok, f16* __restrict__ ov) {
  const int which = blockIdx.y;
  const float* src = which == 0 ? Wq : which == 1 ? Wk : Wv;
  f16* dst = which == 0 ? oq : which == 1 ? ok : ov;
  size_t i = (size_t)blockIdx.x * blockDim.x + threadIdx.x;   // in float4 units
  float4 v = ((const float4*)src)[i];
  union { f16 h[4]; uint2 u; } w;
  w.h[0] = (f16)v.x; w.h[1] = (f16)v.y; w.h[2] = (f16)v.z; w.h[3] = (f16)v.w;
  ((uint2*)dst)[i] = w.u;
}

// ------------------------------------- 3) projection GEMM + head split ------
// out[b,h,n,d] (f16) = (A[row,:] @ W)[col] + bias[col], A: 4096x1024 f16.
// Block tile 64x64, 8 waves each computing 16x32.

__global__ __launch_bounds__(256)
void proj_kernel(const f16* __restrict__ A, const f16* __restrict__ W,
                 const float* __restrict__ bias, f16* __restrict__ out) {
  __shared__ __align__(16) f16 wT[64 * 32];   // [n_local][k_local] transposed tile
  const int lane = threadIdx.x & 31;
  const int wave = threadIdx.x >> 5;
  const int mb = blockIdx.x * 64;
  const int cb = blockIdx.y * 64;
  const int mbase = mb + (wave >> 1) * 16;
  const int cwave = (wave & 1) * 32;          // col offset within block tile

  v8f c0 = {}, c1 = {};
  for (int kb = 0; kb < Emb; kb += 32) {
    {   // stage W[kb..kb+31][cb..cb+63] transposed into LDS (coalesced reads)
      int t  = threadIdx.x;
      int kk = t >> 3;                 // 0..31
      int nl = (t & 7) * 8;            // 0..56
      union { uint4 q; f16 h[8]; } u;
      u.q = *(const uint4*)(W + (size_t)(kb + kk) * Emb + cb + nl);
#pragma unroll
      for (int i = 0; i < 8; ++i) wT[(nl + i) * 32 + kk] = u.h[i];
    }
    __syncthreads();
    v16h a  = load_afrag(A + (size_t)mbase * Emb + kb, Emb, lane);
    v16h b0 = load_bfrag(wT + (size_t)(cwave + 0)  * 32, 32, lane);
    v16h b1 = load_bfrag(wT + (size_t)(cwave + 16) * 32, 32, lane);
    c0 = wmma_f16(a, b0, c0);
    c1 = wmma_f16(a, b1, c1);
    __syncthreads();
  }

  const int mloc = (lane >> 4) << 3;
#pragma unroll
  for (int r = 0; r < 8; ++r) {
    int m = mbase + r + mloc;
    int bbatch = m >> 11;              // /Nseq
    int nseq   = m & (Nseq - 1);
#pragma unroll
    for (int t = 0; t < 2; ++t) {
      int col = cb + cwave + t * 16 + (lane & 15);
      float v = (t ? c1[r] : c0[r]) + bias[col];
      int hh = col >> 6, d = col & 63;
      out[(((size_t)(bbatch * Hh + hh) * Nseq) + nseq) * Dh + d] = (f16)v;
    }
  }
}

// ------------------------------------- 4) causal flash attention ------------
// Grid: (Nseq/64, B*H). Block: 128 (4 waves, 16 query rows per wave).

__global__ __launch_bounds__(128)
void attn_kernel(const f16* __restrict__ qh, const f16* __restrict__ kh,
                 const f16* __restrict__ vh, const float* __restrict__ Qin,
                 float* __restrict__ out) {
  __shared__ __align__(16) f16 ktile[32 * 64];      // [key][d]
  __shared__ __align__(16) f16 vT[64 * 32];         // [d][key]
  __shared__ __align__(16) f16 pbuf[4][16 * 32];    // per-wave P staging

  const int lane = threadIdx.x & 31;
  const int wave = threadIdx.x >> 5;
  const int bh = blockIdx.y;
  const int bbatch = bh >> 4;          // /Hh
  const int head   = bh & (Hh - 1);
  const int rowblk = blockIdx.x * 64;
  const int rowbase = rowblk + wave * 16;
  const size_t headoff = (size_t)bh * Nseq * Dh;

  v16h qf0 = load_afrag(qh + headoff + (size_t)rowbase * Dh + 0,  Dh, lane);
  v16h qf1 = load_afrag(qh + headoff + (size_t)rowbase * Dh + 32, Dh, lane);

  float mrow[8], lrow[8];
#pragma unroll
  for (int r = 0; r < 8; ++r) { mrow[r] = -INFINITY; lrow[r] = 0.0f; }
  v8f o0 = {}, o1 = {}, o2 = {}, o3 = {};

  const int jEnd = rowblk + 64;
  for (int jb = 0; jb < jEnd; jb += 32) {
    {   // stage K tile [32 keys][64 d] via async global->LDS DMA (ASYNCcnt)
      int t = threadIdx.x;
      int key = t >> 2, dl = (t & 3) * 16;
      const f16* src = kh + headoff + (size_t)(jb + key) * Dh + dl;
      f16* dst = &ktile[key * 64 + dl];
      async_copy_b128(dst, src);
      async_copy_b128(dst + 8, src + 8);
    }
    {   // stage V tile transposed: vT[d][key] (transpose -> must go via VGPRs)
      int t = threadIdx.x;
#pragma unroll
      for (int half = 0; half < 2; ++half) {
        int lin = t * 16 + half * 8;          // 0..2047
        int key = lin >> 6, dl = lin & 63;
        union { uint4 q; f16 h[8]; } u;
        u.q = *(const uint4*)(vh + headoff + (size_t)(jb + key) * Dh + dl);
#pragma unroll
        for (int i = 0; i < 8; ++i) vT[(dl + i) * 32 + key] = u.h[i];
      }
    }
    if (jb + 32 < jEnd)   // prefetch next K tile
      __builtin_prefetch(kh + headoff + (size_t)(jb + 32) * Dh, 0, 1);
    async_wait0();        // K-tile async copies complete (LDS written)
    __syncthreads();

    if (jb <= rowbase + 15) {          // tile has at least one unmasked key
      // S = (Q/sqrt(D)) K^T : two 16x16 f32 tiles over 32 keys
      v8f s0 = {}, s1 = {};
      v16h bk00 = load_bfrag(ktile + 0,            64, lane);  // keys 0-15,  d 0-31
      v16h bk01 = load_bfrag(ktile + 32,           64, lane);  // keys 0-15,  d 32-63
      v16h bk10 = load_bfrag(ktile + 16 * 64,      64, lane);  // keys 16-31, d 0-31
      v16h bk11 = load_bfrag(ktile + 16 * 64 + 32, 64, lane);
      s0 = wmma_f16(qf0, bk00, s0); s0 = wmma_f16(qf1, bk01, s0);
      s1 = wmma_f16(qf0, bk10, s1); s1 = wmma_f16(qf1, bk11, s1);

      const int mloc = (lane >> 4) << 3;
      const int n0 = jb + (lane & 15), n1 = n0 + 16;
#pragma unroll
      for (int r = 0; r < 8; ++r) {
        int m = rowbase + r + mloc;
        float x0 = (n0 <= m) ? s0[r] * 0.125f : -INFINITY;   // 1/sqrt(64)
        float x1 = (n1 <= m) ? s1[r] * 0.125f : -INFINITY;
        float rmax = rowmax16(fmaxf(x0, x1));
        float mnew = fmaxf(mrow[r], rmax);
        float corr = __expf(mrow[r] - mnew);
        float p0 = __expf(x0 - mnew);
        float p1 = __expf(x1 - mnew);
        lrow[r] = lrow[r] * corr + rowsum16(p0 + p1);
        mrow[r] = mnew;
        o0[r] *= corr; o1[r] *= corr; o2[r] *= corr; o3[r] *= corr;
        f16* pw = &pbuf[wave][(r + mloc) * 32 + (lane & 15)];
        pw[0]  = (f16)p0;
        pw[16] = (f16)p1;
      }
      wave_lds_fence();
      // O += P(16x32) * V(32x64)
      v16h pf  = load_afrag(&pbuf[wave][0], 32, lane);
      v16h bv0 = load_bfrag(vT + 0 * 16 * 32, 32, lane);
      v16h bv1 = load_bfrag(vT + 1 * 16 * 32, 32, lane);
      v16h bv2 = load_bfrag(vT + 2 * 16 * 32, 32, lane);
      v16h bv3 = load_bfrag(vT + 3 * 16 * 32, 32, lane);
      o0 = wmma_f16(pf, bv0, o0);
      o1 = wmma_f16(pf, bv1, o1);
      o2 = wmma_f16(pf, bv2, o2);
      o3 = wmma_f16(pf, bv3, o3);
    }
    __syncthreads();
  }

  // epilogue: O / l + residual
  const int mloc = (lane >> 4) << 3;
#pragma unroll
  for (int r = 0; r < 8; ++r) {
    int m = rowbase + r + mloc;
    float inv = 1.0f / lrow[r];
    size_t rowoff = ((size_t)bbatch * Nseq + m) * Emb + head * Dh;
    int c = lane & 15;
    out[rowoff +  0 + c] = o0[r] * inv + Qin[rowoff +  0 + c];
    out[rowoff + 16 + c] = o1[r] * inv + Qin[rowoff + 16 + c];
    out[rowoff + 32 + c] = o2[r] * inv + Qin[rowoff + 32 + c];
    out[rowoff + 48 + c] = o3[r] * inv + Qin[rowoff + 48 + c];
  }
}

// ---------------------------------------------------------------- launch ----

extern "C" void kernel_launch(void* const* d_in, const int* in_sizes, int n_in,
                              void* d_out, int out_size, void* d_ws, size_t ws_size,
                              hipStream_t stream) {
  const float* Q     = (const float*)d_in[0];
  const float* K     = (const float*)d_in[1];
  const float* V     = (const float*)d_in[2];
  // d_in[3] = causal mask (recomputed analytically)
  const float* Wq    = (const float*)d_in[4];
  const float* bq    = (const float*)d_in[5];
  const float* Wk    = (const float*)d_in[6];
  const float* bk    = (const float*)d_in[7];
  const float* Wv    = (const float*)d_in[8];
  const float* bv    = (const float*)d_in[9];
  const float* gamma = (const float*)d_in[10];
  const float* beta  = (const float*)d_in[11];
  float* out = (float*)d_out;

  const size_t ACT = (size_t)Bdim * Nseq * Emb;   // 4M halves per activation buffer
  const size_t WEL = (size_t)Emb * Emb;
  f16* ws  = (f16*)d_ws;
  f16* lnQ = ws;            f16* lnK = lnQ + ACT;  f16* lnV = lnK + ACT;
  f16* wq16 = lnV + ACT;    f16* wk16 = wq16 + WEL; f16* wv16 = wk16 + WEL;
  f16* qh = wv16 + WEL;     f16* khp = qh + ACT;    f16* vhp = khp + ACT;

  ln3_kernel<<<dim3(Bdim * Nseq, 3), 256, 0, stream>>>(Q, K, V, gamma, beta, lnQ, lnK, lnV);
  cvtw_kernel<<<dim3((unsigned)(WEL / (256 * 4)), 3), 256, 0, stream>>>(Wq, Wk, Wv, wq16, wk16, wv16);

  dim3 pgrid(Bdim * Nseq / 64, Emb / 64);
  proj_kernel<<<pgrid, 256, 0, stream>>>(lnQ, wq16, bq, qh);
  proj_kernel<<<pgrid, 256, 0, stream>>>(lnK, wk16, bk, khp);
  proj_kernel<<<pgrid, 256, 0, stream>>>(lnV, wv16, bv, vhp);

  attn_kernel<<<dim3(Nseq / 64, Bdim * Hh), 128, 0, stream>>>(qh, khp, vhp, Q, out);
}
